// SNN_17660905521720
// MI455X (gfx1250) — matrix operand order repaired and checked by reading, not compiled, not run on hardware
//
#include <hip/hip_runtime.h>
#include <stdint.h>

// ---------------------------------------------------------------------------
// SNN forward: GEMM1(bf16 WMMA) -> LIF1 scan -> GEMM2(bf16 WMMA) -> LIF2 scan
// Shapes fixed: T=500 B=128 NI=784 NH=512 NO=10
// d_out layout (floats): spk1[500*128*512] | trace[500] | spk2[500*128*10] | mem2[500*128*10]
// ---------------------------------------------------------------------------

typedef __bf16 bf16_t;
typedef bf16_t bf16x16 __attribute__((ext_vector_type(16)));
typedef float  f32x8   __attribute__((ext_vector_type(8)));

union FragBF { bf16x16 v; uint32_t u[8]; };

__device__ __forceinline__ uint32_t pack_bf16(float a, float b) {
    uint32_t ua = __float_as_uint(a), ub = __float_as_uint(b);
    ua += 0x7FFFu + ((ua >> 16) & 1u);   // round-to-nearest-even
    ub += 0x7FFFu + ((ub >> 16) & 1u);
    return (ua >> 16) | (ub & 0xFFFF0000u);
}

#define SNN_T   500
#define SNN_B   128
#define SNN_NI  784
#define SNN_NH  512
#define SNN_NO  10
#define ROWS    (SNN_T * SNN_B)          // 64000
#define CK      160                      // K-chunk (multiple of 32, covers 784 in 5 chunks)
#define CKD     (CK / 2)                 // dwords per row per chunk

// ---------------------------------------------------------------------------
// Kernel 1: cur1[row,h] = sum_i x[row,i]*W1[h,i] + b1[h] + 0.2
// Block: 256 thr (8 waves). Tile M=32, N=128. Each wave: 1 m-tile x 2 n-tiles.
// Grid: (ROWS/32, NH/128) = (2000, 4)
// ---------------------------------------------------------------------------
__global__ __launch_bounds__(256) void snn_gemm1(
    const float* __restrict__ x, const float* __restrict__ W1,
    const float* __restrict__ b1, float* __restrict__ cur1)
{
    __shared__ uint32_t sA[32 * CKD];    // 32 rows x 160 bf16  (10.2 KB)
    __shared__ uint32_t sB[128 * CKD];   // 128 rows x 160 bf16 (40.9 KB)

    const int tid  = threadIdx.x;
    const int lane = tid & 31;
    const int wave = tid >> 5;
    const int mbase = blockIdx.x * 32;
    const int hbase = blockIdx.y * 128;
    const int mt  = wave >> 2;           // 0..1
    const int nt0 = (wave & 3) * 2;      // 0,2,4,6
    const int l15 = lane & 15;

    f32x8 acc0 = {0.f,0.f,0.f,0.f,0.f,0.f,0.f,0.f};
    f32x8 acc1 = {0.f,0.f,0.f,0.f,0.f,0.f,0.f,0.f};

    for (int c = 0; c < 5; ++c) {
        const int kc = c * CK;
        // stage A chunk (x -> bf16)
        for (int i = tid; i < 32 * CKD; i += 256) {
            const int r = i / CKD, cd = i % CKD;
            const int k = kc + cd * 2;
            const float* xr = x + (size_t)(mbase + r) * SNN_NI;
            const float f0 = (k     < SNN_NI) ? xr[k]     : 0.f;
            const float f1 = (k + 1 < SNN_NI) ? xr[k + 1] : 0.f;
            sA[i] = pack_bf16(f0, f1);
        }
        // stage B chunk (W1 rows -> bf16); B[k][n] = W1[hbase+n][k]
        for (int i = tid; i < 128 * CKD; i += 256) {
            const int n = i / CKD, cd = i % CKD;
            const int k = kc + cd * 2;
            const float* wr = W1 + (size_t)(hbase + n) * SNN_NI;
            const float f0 = (k     < SNN_NI) ? wr[k]     : 0.f;
            const float f1 = (k + 1 < SNN_NI) ? wr[k + 1] : 0.f;
            sB[i] = pack_bf16(f0, f1);
        }
        __syncthreads();

        const uint32_t* aRow  = sA + (mt * 16 + l15) * CKD;
        const uint32_t* bRow0 = sB + ((nt0 + 0) * 16 + l15) * CKD;
        const uint32_t* bRow1 = sB + ((nt0 + 1) * 16 + l15) * CKD;

        #pragma unroll
        for (int ks = 0; ks < 5; ++ks) {
            const int k0d = ks * 16;                      // k0/2 in dwords
            FragBF a, b0, b1f;
            // A k order used here (same as B): lanes<16 hold K 0..15; lanes>=16 hold K 16..31
            const int alo = k0d + ((lane < 16) ? 0 : 8);
            const int ahi = k0d + ((lane < 16) ? 4 : 12);
            #pragma unroll
            for (int j = 0; j < 4; ++j) { a.u[j] = aRow[alo + j]; a.u[4 + j] = aRow[ahi + j]; }
            // B k order used here (same as A): lanes<16 hold K 0..15; lanes>=16 hold K 16..31 (col = lane%16)
            const int boff = k0d + ((lane < 16) ? 0 : 8);
            #pragma unroll
            for (int j = 0; j < 8; ++j) { b0.u[j] = bRow0[boff + j]; b1f.u[j] = bRow1[boff + j]; }

            acc0 = __builtin_amdgcn_wmma_f32_16x16x32_bf16(false, a.v, false, b0.v,
                                                           (short)0, acc0, false, false);
            acc1 = __builtin_amdgcn_wmma_f32_16x16x32_bf16(false, a.v, false, b1f.v,
                                                           (short)0, acc1, false, false);
        }
        __syncthreads();
    }

    // epilogue: + b1 + 0.2, store f32 tile. C layout: col = lane%16, row r (lanes<16) / 8+r
    const int n0 = hbase + (nt0 + 0) * 16 + l15;
    const int n1 = hbase + (nt0 + 1) * 16 + l15;
    const float bias0 = b1[n0] + 0.2f;
    const float bias1 = b1[n1] + 0.2f;
    const int mrow0 = mbase + mt * 16 + ((lane < 16) ? 0 : 8);
    #pragma unroll
    for (int r = 0; r < 8; ++r) {
        cur1[(size_t)(mrow0 + r) * SNN_NH + n0] = acc0[r] + bias0;
        cur1[(size_t)(mrow0 + r) * SNN_NH + n1] = acc1[r] + bias1;
    }
}

// ---------------------------------------------------------------------------
// Kernel 2: LIF layer-1 scan, in place cur1 -> spk1. One thread per (b,h).
// ---------------------------------------------------------------------------
__global__ __launch_bounds__(256) void snn_lif1(float* __restrict__ buf,
                                                float* __restrict__ trace)
{
    const int idx = blockIdx.x * 256 + threadIdx.x;       // 0..65535 = b*512+h
    float mem = 0.f;
    float* p = buf + idx;
    const bool isTrace = (idx == 0);                       // (b=0,h=0)
    for (int t = 0; t < SNN_T; ++t) {
        const float cur   = *p;
        const float reset = (mem > 1.f) ? 1.f : 0.f;       // from PREVIOUS mem
        mem = 0.9f * mem + cur - reset;
        *p = (mem > 1.f) ? 1.f : 0.f;                      // spike (mem-1 > 0)
        if (isTrace) trace[t] = mem;
        p += SNN_B * SNN_NH;
    }
}

// ---------------------------------------------------------------------------
// Kernel 3: cur2[row,o] = sum_h spk1[row,h]*W2[o,h] + b2[o]   (N padded 10->16)
// Block: 256 thr (8 waves), each wave a 16-row m-tile -> 128 rows/block.
// Grid: ROWS/128 = 500
// ---------------------------------------------------------------------------
__global__ __launch_bounds__(256) void snn_gemm2(
    const float* __restrict__ spk1, const float* __restrict__ W2,
    const float* __restrict__ b2, float* __restrict__ cur2)
{
    __shared__ uint32_t sW[16 * 256];    // 16 x 512 bf16 (16 KB), rows 10..15 zero
    const int tid = threadIdx.x, lane = tid & 31, wave = tid >> 5;
    const int l15 = lane & 15;

    for (int i = tid; i < 16 * 256; i += 256) {
        const int n = i >> 8, cd = i & 255;
        const int k = cd * 2;
        const float f0 = (n < SNN_NO) ? W2[n * SNN_NH + k]     : 0.f;
        const float f1 = (n < SNN_NO) ? W2[n * SNN_NH + k + 1] : 0.f;
        sW[i] = pack_bf16(f0, f1);
    }
    __syncthreads();

    const int rowbase = blockIdx.x * 128 + wave * 16;
    const float* arow = spk1 + (size_t)(rowbase + l15) * SNN_NH;  // this lane's A row
    const uint32_t* bRow = sW + l15 * 256;

    f32x8 acc = {0.f,0.f,0.f,0.f,0.f,0.f,0.f,0.f};
    #pragma unroll 4
    for (int ks = 0; ks < 16; ++ks) {
        const int k0 = ks * 32;
        FragBF a, b;
        const int klo = k0 + ((lane < 16) ? 0 : 16);
        const int khi = k0 + ((lane < 16) ? 8 : 24);
        #pragma unroll
        for (int j = 0; j < 4; ++j) {
            a.u[j]     = pack_bf16(arow[klo + 2 * j], arow[klo + 2 * j + 1]); // exact: 0/1
            a.u[4 + j] = pack_bf16(arow[khi + 2 * j], arow[khi + 2 * j + 1]);
        }
        const int boff = (k0 >> 1) + ((lane < 16) ? 0 : 8);
        #pragma unroll
        for (int j = 0; j < 8; ++j) b.u[j] = bRow[boff + j];
        acc = __builtin_amdgcn_wmma_f32_16x16x32_bf16(false, a.v, false, b.v,
                                                      (short)0, acc, false, false);
    }

    const int n = l15;
    if (n < SNN_NO) {
        const float bias = b2[n];
        const int mrow0 = rowbase + ((lane < 16) ? 0 : 8);
        #pragma unroll
        for (int r = 0; r < 8; ++r)
            cur2[(size_t)(mrow0 + r) * SNN_NO + n] = acc[r] + bias;
    }
}

// ---------------------------------------------------------------------------
// Kernel 4: LIF layer-2 scan, in place cur2 -> spk2, plus mem2_rec.
// ---------------------------------------------------------------------------
__global__ __launch_bounds__(128) void snn_lif2(float* __restrict__ buf,
                                                float* __restrict__ memrec)
{
    const int idx = blockIdx.x * 128 + threadIdx.x;       // 0..1279 = b*10+o
    if (idx >= SNN_B * SNN_NO) return;
    float mem = 0.f;
    float* p = buf + idx;
    float* q = memrec + idx;
    for (int t = 0; t < SNN_T; ++t) {
        const float cur   = *p;
        const float reset = (mem > 1.f) ? 1.f : 0.f;
        mem = 0.8f * mem + cur - reset;
        *q = mem;
        *p = (mem > 1.f) ? 1.f : 0.f;
        p += SNN_B * SNN_NO;
        q += SNN_B * SNN_NO;
    }
}

// ---------------------------------------------------------------------------
extern "C" void kernel_launch(void* const* d_in, const int* in_sizes, int n_in,
                              void* d_out, int out_size, void* d_ws, size_t ws_size,
                              hipStream_t stream) {
    (void)in_sizes; (void)n_in; (void)out_size; (void)d_ws; (void)ws_size;
    const float* x  = (const float*)d_in[0];
    const float* W1 = (const float*)d_in[1];
    const float* b1 = (const float*)d_in[2];
    const float* W2 = (const float*)d_in[3];
    const float* b2 = (const float*)d_in[4];

    float* out   = (float*)d_out;
    float* spk1  = out;                                    // staged as cur1 first
    float* trace = out + (size_t)SNN_T * SNN_B * SNN_NH;   // +32768000
    float* spk2  = trace + SNN_T;                          // staged as cur2 first
    float* mem2  = spk2 + (size_t)SNN_T * SNN_B * SNN_NO;  // +640000

    snn_gemm1<<<dim3(ROWS / 32, SNN_NH / 128), 256, 0, stream>>>(x, W1, b1, spk1);
    snn_lif1 <<<(SNN_B * SNN_NH) / 256, 256, 0, stream>>>(spk1, trace);
    snn_gemm2<<<ROWS / 128, 256, 0, stream>>>(spk1, W2, b2, spk2);
    snn_lif2 <<<(SNN_B * SNN_NO + 127) / 128, 128, 0, stream>>>(spk2, mem2);
}
